// GIN_LCG_15839839387883
// MI455X (gfx1250) — matrix operand order repaired and checked by reading, not compiled, not run on hardware
//
#include <hip/hip_runtime.h>

// ---------------------------------------------------------------------------
// CDNA5 / gfx1250 implementation of the literal-clause GNN.
//  - All 5 MLPs -> one generic 2-layer WMMA kernel (bf16 A/B, f32 accum).
//  - Edge gather + segment-sum fused into one atomic scatter kernel.
//  - Messages materialized per-node (L2-resident), never per-edge.
//  - 2 row sub-tiles per block to amortize LDS weight staging; prefetch of
//    the next sub-tile's activations (global_prefetch_b8).
// ---------------------------------------------------------------------------

typedef __attribute__((ext_vector_type(16))) __bf16 v16bf;
typedef __attribute__((ext_vector_type(8)))  __bf16 v8bf;
typedef __attribute__((ext_vector_type(8)))  float  v8f;

#define KPAD 200   // padded K stride for transposed W1 in LDS (multiple of 8)
#define HPAD 72    // padded row stride for hidden / W2 tiles (multiple of 8)
#define SUBT 2     // 128-row sub-tiles per block (weights staged once)

static __device__ __forceinline__ v8f wmma_bf16(v16bf a, v16bf b, v8f c) {
  // D = A(16x32 bf16) * B(32x16 bf16) + C(16x16 f32)
  return __builtin_amdgcn_wmma_f32_16x16x32_bf16(
      /*neg_a=*/false, a, /*neg_b=*/false, b,
      /*c_mod=*/(short)0, c, /*reuse_a=*/false, /*reuse_b=*/false);
}

static __device__ __forceinline__ v16bf cat8(v8bf lo, v8bf hi) {
  v16bf r;
#pragma unroll
  for (int j = 0; j < 8; ++j) { r[j] = lo[j]; r[j + 8] = hi[j]; }
  return r;
}

// Generic fused 2-layer MLP:  Y = relu(X @ W1 + b1) @ W2 + b2
// X is the virtual concat of up to 3 sources, each 64 columns wide.
// K = nseg*64 (64 / 128 / 192).  flip: row r reads source row r^1 (X0 only).
__global__ __launch_bounds__(256, 2) void mlp2_wmma_kernel(
    const float* __restrict__ X0, const float* __restrict__ X1,
    const float* __restrict__ X2, int nseg,
    const float* __restrict__ W1, const float* __restrict__ b1,
    const float* __restrict__ W2, const float* __restrict__ b2,
    float* __restrict__ Y, int N, int flip)
{
  __shared__ __align__(16) __bf16 Wt [64][KPAD];  // W1 transposed: Wt[col][k]
  __shared__ __align__(16) __bf16 W2t[64][HPAD];  // W2 transposed
  __shared__ float bb1[64];
  __shared__ float bb2[64];
  __shared__ __align__(16) __bf16 hs[8][16][HPAD]; // per-wave relu(hidden) tile

  const int tid = threadIdx.x;
  const int K   = nseg << 6;

  // Stage weights (transposed, bf16) + biases (f32) into LDS once per block.
  for (int idx = tid; idx < (K << 6); idx += 256)
    Wt[idx & 63][idx >> 6] = (__bf16)W1[idx];
  for (int idx = tid; idx < 4096; idx += 256)
    W2t[idx & 63][idx >> 6] = (__bf16)W2[idx];
  if (tid < 64) { bb1[tid] = b1[tid]; bb2[tid] = b2[tid]; }
  __syncthreads();

  const int wave = tid >> 5;
  const int lane = tid & 31;
  const int half = lane >> 4;   // 0: lanes 0-15, 1: lanes 16-31
  const int ln   = lane & 15;
  const int row0 = blockIdx.x * (128 * SUBT);

  // Prefetch the second sub-tile's activation rows while sub-tile 0 computes.
  {
    const int prow = row0 + 128 + wave * 16 + ln;
    if (prow < N) __builtin_prefetch(X0 + (size_t)prow * 64, 0, 0);
  }

  for (int sub = 0; sub < SUBT; ++sub) {
    const int row_base = row0 + sub * 128 + wave * 16;
    if (row_base >= N) break;   // wave-uniform (after the block barrier)

    int arow = row_base + ln;
    if (flip) arow ^= 1;        // literal <-> negation pair swap

    // ---- Layer 1: acc(16x64) = A(16xK) x W1(Kx64), K stepped by 32 ----
    v8f acc0 = {}, acc1 = {}, acc2 = {}, acc3 = {};
    for (int k0 = 0; k0 < K; k0 += 32) {
      // A fragment (ISA 16-bit A 16x32 layout): lane holds row (lane&15),
      // cols [k0+half*8, +8) then [k0+16+half*8, +8).
      const int c0 = k0 + half * 8;
      const int c1 = c0 + 16;
      const float* p0 = (c0 < 64 ? X0 : (c0 < 128 ? X1 : X2)) + (size_t)arow * 64 + (c0 & 63);
      const float* p1 = (c1 < 64 ? X0 : (c1 < 128 ? X1 : X2)) + (size_t)arow * 64 + (c1 & 63);
      const float4 f0 = *(const float4*)p0;
      const float4 f1 = *(const float4*)(p0 + 4);
      const float4 f2 = *(const float4*)p1;
      const float4 f3 = *(const float4*)(p1 + 4);
      v16bf a;
      a[0]=(__bf16)f0.x;  a[1]=(__bf16)f0.y;  a[2]=(__bf16)f0.z;  a[3]=(__bf16)f0.w;
      a[4]=(__bf16)f1.x;  a[5]=(__bf16)f1.y;  a[6]=(__bf16)f1.z;  a[7]=(__bf16)f1.w;
      a[8]=(__bf16)f2.x;  a[9]=(__bf16)f2.y;  a[10]=(__bf16)f2.z; a[11]=(__bf16)f2.w;
      a[12]=(__bf16)f3.x; a[13]=(__bf16)f3.y; a[14]=(__bf16)f3.z; a[15]=(__bf16)f3.w;

      // B fragments: lane = col n, K values [k0+half*8, +8) then
      // [k0+16+half*8, +8) (transposed-in-LDS makes this two ds_load_b128).
      const int kb = k0 + half * 8;
      v16bf B0 = cat8(*(const v8bf*)&Wt[ 0 + ln][kb], *(const v8bf*)&Wt[ 0 + ln][kb + 16]);
      v16bf B1 = cat8(*(const v8bf*)&Wt[16 + ln][kb], *(const v8bf*)&Wt[16 + ln][kb + 16]);
      v16bf B2 = cat8(*(const v8bf*)&Wt[32 + ln][kb], *(const v8bf*)&Wt[32 + ln][kb + 16]);
      v16bf B3 = cat8(*(const v8bf*)&Wt[48 + ln][kb], *(const v8bf*)&Wt[48 + ln][kb + 16]);
      acc0 = wmma_bf16(a, B0, acc0);
      acc1 = wmma_bf16(a, B1, acc1);
      acc2 = wmma_bf16(a, B2, acc2);
      acc3 = wmma_bf16(a, B3, acc3);
    }

    // ---- bias + ReLU -> bf16 hidden tile in wave-private LDS ----
    // D layout: VGPR r, lanes 0-15 -> row r, lanes 16-31 -> row r+8; col = ln.
    const int m0 = half * 8;
#pragma unroll
    for (int r = 0; r < 8; ++r) {
      hs[wave][m0 + r][ 0 + ln] = (__bf16)fmaxf(acc0[r] + bb1[ 0 + ln], 0.0f);
      hs[wave][m0 + r][16 + ln] = (__bf16)fmaxf(acc1[r] + bb1[16 + ln], 0.0f);
      hs[wave][m0 + r][32 + ln] = (__bf16)fmaxf(acc2[r] + bb1[32 + ln], 0.0f);
      hs[wave][m0 + r][48 + ln] = (__bf16)fmaxf(acc3[r] + bb1[48 + ln], 0.0f);
    }
    // Wave-private region; DS ops are in-order for the wave -> no barrier.

    // ---- Layer 2: out(16x64) = relu(H)(16x64) x W2(64x64) ----
    v8f o0 = {}, o1 = {}, o2 = {}, o3 = {};
#pragma unroll
    for (int k0 = 0; k0 < 64; k0 += 32) {
      const __bf16* hrow = &hs[wave][ln][0];
      v16bf a = cat8(*(const v8bf*)(hrow + k0 + half * 8),
                     *(const v8bf*)(hrow + k0 + 16 + half * 8));
      const int kb = k0 + half * 8;
      v16bf B0 = cat8(*(const v8bf*)&W2t[ 0 + ln][kb], *(const v8bf*)&W2t[ 0 + ln][kb + 16]);
      v16bf B1 = cat8(*(const v8bf*)&W2t[16 + ln][kb], *(const v8bf*)&W2t[16 + ln][kb + 16]);
      v16bf B2 = cat8(*(const v8bf*)&W2t[32 + ln][kb], *(const v8bf*)&W2t[32 + ln][kb + 16]);
      v16bf B3 = cat8(*(const v8bf*)&W2t[48 + ln][kb], *(const v8bf*)&W2t[48 + ln][kb + 16]);
      o0 = wmma_bf16(a, B0, o0);
      o1 = wmma_bf16(a, B1, o1);
      o2 = wmma_bf16(a, B2, o2);
      o3 = wmma_bf16(a, B3, o3);
    }

#pragma unroll
    for (int r = 0; r < 8; ++r) {
      const size_t yo = (size_t)(row_base + m0 + r) * 64;
      Y[yo +  0 + ln] = o0[r] + bb2[ 0 + ln];
      Y[yo + 16 + ln] = o1[r] + bb2[16 + ln];
      Y[yo + 32 + ln] = o2[r] + bb2[32 + ln];
      Y[yo + 48 + ln] = o3[r] + bb2[48 + ln];
    }
  }
}

// Fused gather + segment-sum:  aggr[dst[e]] += msg[src[e]]  (64 f32 per edge).
// msg / aggr arrays fit in the 192MB L2, so atomics stay on-chip.
__global__ __launch_bounds__(256) void scatter_add_kernel(
    const float* __restrict__ msg, const int* __restrict__ src,
    const int* __restrict__ dst, float* __restrict__ aggr, int E)
{
  const size_t tid = (size_t)blockIdx.x * 256 + threadIdx.x;
  if (tid >= (size_t)E * 32) return;
  const int e = (int)(tid >> 5);
  const int c = ((int)tid & 31) * 2;
  const int s = src[e];
  const int d = dst[e];
  const float2 v = *(const float2*)(msg + (size_t)s * 64 + c);
  float* ap = aggr + (size_t)d * 64 + c;
  atomicAdd(ap,     v.x);
  atomicAdd(ap + 1, v.y);
}

__global__ void zero_kernel(float* __restrict__ p, size_t n) {
  size_t i = (size_t)blockIdx.x * blockDim.x + threadIdx.x;
  const size_t stride = (size_t)gridDim.x * blockDim.x;
  for (; i < n; i += stride) p[i] = 0.0f;
}

extern "C" void kernel_launch(void* const* d_in, const int* in_sizes, int n_in,
                              void* d_out, int out_size, void* d_ws, size_t ws_size,
                              hipStream_t stream)
{
  const int*   l_edge = (const int*)d_in[2];
  const int*   c_edge = (const int*)d_in[3];
  const float* l_emb0 = (const float*)d_in[4];
  const float* c_emb0 = (const float*)d_in[5];
  const int L = in_sizes[4] / 64;
  const int C = in_sizes[5] / 64;
  const int E = in_sizes[2];

  const float* l2c_W1 = (const float*)d_in[6];
  const float* l2c_b1 = (const float*)d_in[7];
  const float* l2c_W2 = (const float*)d_in[8];
  const float* l2c_b2 = (const float*)d_in[9];
  const float* c2l_W1 = (const float*)d_in[10];
  const float* c2l_b1 = (const float*)d_in[11];
  const float* c2l_W2 = (const float*)d_in[12];
  const float* c2l_b2 = (const float*)d_in[13];
  const float* l2l_W1 = (const float*)d_in[14];
  const float* l2l_b1 = (const float*)d_in[15];
  const float* l2l_W2 = (const float*)d_in[16];
  const float* l2l_b2 = (const float*)d_in[17];
  const float* cu_W1  = (const float*)d_in[18];
  const float* cu_b1  = (const float*)d_in[19];
  const float* cu_W2  = (const float*)d_in[20];
  const float* cu_b2  = (const float*)d_in[21];
  const float* lu_W1  = (const float*)d_in[22];
  const float* lu_b1  = (const float*)d_in[23];
  const float* lu_W2  = (const float*)d_in[24];
  const float* lu_b2  = (const float*)d_in[25];

  // Output layout: l_embs stack [3][L][64] then c_embs stack [3][C][64].
  // Iteration-i embeddings live directly in their output slices.
  float* outL = (float*)d_out;
  float* outC = outL + (size_t)3 * L * 64;

  float* ws = (float*)d_ws;
  float* l2c_msg  = ws; ws += (size_t)L * 64;
  float* c2l_msg  = ws; ws += (size_t)C * 64;
  float* l2l_msg  = ws; ws += (size_t)L * 64;
  float* l2c_aggr = ws; ws += (size_t)C * 64;
  float* c2l_aggr = ws; ws += (size_t)L * 64;

  hipMemcpyAsync(outL, l_emb0, (size_t)L * 64 * sizeof(float),
                 hipMemcpyDeviceToDevice, stream);
  hipMemcpyAsync(outC, c_emb0, (size_t)C * 64 * sizeof(float),
                 hipMemcpyDeviceToDevice, stream);

  const dim3 blk(256);
  const dim3 gL((L + 128 * SUBT - 1) / (128 * SUBT));
  const dim3 gC((C + 128 * SUBT - 1) / (128 * SUBT));
  const dim3 gS((unsigned)(((size_t)E * 32 + 255) / 256));

  for (int it = 0; it < 2; ++it) {
    const float* lcur = outL + (size_t)it * L * 64;
    const float* ccur = outC + (size_t)it * C * 64;
    float* lnext = outL + (size_t)(it + 1) * L * 64;
    float* cnext = outC + (size_t)(it + 1) * C * 64;

    // Node-level messages (computed once per node, gathered per edge later).
    mlp2_wmma_kernel<<<gL, blk, 0, stream>>>(lcur, nullptr, nullptr, 1,
        l2c_W1, l2c_b1, l2c_W2, l2c_b2, l2c_msg, L, 0);
    mlp2_wmma_kernel<<<gC, blk, 0, stream>>>(ccur, nullptr, nullptr, 1,
        c2l_W1, c2l_b1, c2l_W2, c2l_b2, c2l_msg, C, 0);
    mlp2_wmma_kernel<<<gL, blk, 0, stream>>>(lcur, nullptr, nullptr, 1,
        l2l_W1, l2l_b1, l2l_W2, l2l_b2, l2l_msg, L, 1 /*flip*/);

    // Clause update: aggr = segsum(l2c_msg over edges); c' = MLP([c, aggr]).
    zero_kernel<<<dim3(2048), blk, 0, stream>>>(l2c_aggr, (size_t)C * 64);
    scatter_add_kernel<<<gS, blk, 0, stream>>>(l2c_msg, l_edge, c_edge, l2c_aggr, E);
    mlp2_wmma_kernel<<<gC, blk, 0, stream>>>(ccur, l2c_aggr, nullptr, 2,
        cu_W1, cu_b1, cu_W2, cu_b2, cnext, C, 0);

    // Literal update: aggr = segsum(c2l_msg); l' = MLP([l, aggr, l2l_msg]).
    zero_kernel<<<dim3(2048), blk, 0, stream>>>(c2l_aggr, (size_t)L * 64);
    scatter_add_kernel<<<gS, blk, 0, stream>>>(c2l_msg, c_edge, l_edge, c2l_aggr, E);
    mlp2_wmma_kernel<<<gL, blk, 0, stream>>>(lcur, c2l_aggr, l2l_msg, 3,
        lu_W1, lu_b1, lu_W2, lu_b2, lnext, L, 0);
  }
}
